// MacheteLayer_62637803045574
// MI455X (gfx1250) — compile-verified, matrix operand order checked by reading, not run on hardware
//
#include <hip/hip_runtime.h>
#include <hip/hip_bf16.h>
#include <stdint.h>

// ---- CDNA5 WMMA vector types (probe-confirmed signatures) ----
typedef __attribute__((ext_vector_type(16))) __bf16 v16bf;
typedef __attribute__((ext_vector_type(8)))  __bf16 v8bf;
typedef __attribute__((ext_vector_type(8)))  float  v8f;

#define M_DIM 4096
#define K_DIM 4096
#define N_DIM 12288
#define GROUP 128

#define BM 128
#define BN 256
#define BK 32
#define LDA 40   // bf16 elems per LDS row: 32 + 8 pad -> 80B stride, conflict-free b128 frag loads
#define LDB 40
#define KTILES (K_DIM / BK)

// Native bf16 conversion (gfx1250 hardware bf16 cvt); pack 2 into a dword.
__device__ __forceinline__ uint32_t pack2_bf16(float lo, float hi) {
  union { __bf16 v[2]; uint32_t u; } p;
  p.v[0] = (__bf16)lo;
  p.v[1] = (__bf16)hi;
  return p.u;
}

// Load one 16x32 bf16 WMMA operand fragment from an LDS row base.
// ISA layout: lane holds row (lane&15); K pairs at k0=(lane>>4)*8 and k0+16.
// Two 16B-aligned ds_load_b128 per fragment (row stride 80B; k offsets 0/16/32/48B).
__device__ __forceinline__ v16bf load_frag(const unsigned short* __restrict__ row_base, int k0) {
  v8bf lo = *reinterpret_cast<const v8bf*>(row_base + k0);
  v8bf hi = *reinterpret_cast<const v8bf*>(row_base + k0 + 16);
  return __builtin_shufflevector(lo, hi, 0, 1, 2, 3, 4, 5, 6, 7,
                                         8, 9, 10, 11, 12, 13, 14, 15);
}

__global__ __launch_bounds__(256)
void machete_w4a32_bf16_wmma(const float* __restrict__ A,    // [M,K] f32
                             const int*   __restrict__ Bq,   // [K,N] int in [0,16)
                             const float* __restrict__ S,    // [K/128, N] f32
                             float*       __restrict__ C) {  // [M,N] f32
  // Ping-pong LDS: one barrier per K-tile.
  __shared__ alignas(16) unsigned short sA [2][BM * LDA];  // 2 x 10.0 KB
  __shared__ alignas(16) unsigned short sBt[2][BN * LDB];  // 2 x 20.0 KB  (total 60 KB)

  const int tid  = threadIdx.x;
  const int lane = tid & 31;
  const int wave = tid >> 5;
  const int wm   = wave & 3;   // 4 waves along M: 32 rows each
  const int wn   = wave >> 2;  // 2 waves along N: 128 cols each

  const int m0 = blockIdx.x * BM;   // gridDim.x = 32
  const int n0 = blockIdx.y * BN;   // gridDim.y = 48

  v8f acc[2][8];
#pragma unroll
  for (int mi = 0; mi < 2; ++mi)
#pragma unroll
    for (int ni = 0; ni < 8; ++ni)
      acc[mi][ni] = (v8f){0.f, 0.f, 0.f, 0.f, 0.f, 0.f, 0.f, 0.f};

  const int frag_r  = lane & 15;          // row within 16x16 subtile
  const int frag_k0 = (lane >> 4) << 3;   // 0 or 8

  // B staging ownership: each thread owns 4 consecutive N columns x 8 K rows.
  const int b_cq = tid & 63;          // column-quad: n = b_cq*4 .. +3 (coalesced 512B/wave loads)
  const int b_ko = (tid >> 6) << 3;   // k-octet: 0/8/16/24

  // ---- stage tile `kt` into LDS buffer `buf` ----
  auto stage = [&](int kt, int buf) {
    const int k0 = kt * BK;

    // A: 128x32 f32 -> bf16.  Thread owns 2 (row, k-octet) segments.
#pragma unroll
    for (int i = 0; i < 2; ++i) {
      const int j   = tid + 256 * i;
      const int row = j >> 2;
      const int oct = (j & 3) << 3;
      const float4 v0 = *reinterpret_cast<const float4*>(
          &A[(size_t)(m0 + row) * K_DIM + k0 + oct]);
      const float4 v1 = *reinterpret_cast<const float4*>(
          &A[(size_t)(m0 + row) * K_DIM + k0 + oct + 4]);
      uint4 u;
      u.x = pack2_bf16(v0.x, v0.y);
      u.y = pack2_bf16(v0.z, v0.w);
      u.z = pack2_bf16(v1.x, v1.y);
      u.w = pack2_bf16(v1.z, v1.w);
      *reinterpret_cast<uint4*>(&sA[buf][row * LDA + oct]) = u;  // ds_store_b128
    }

    // B: 32x256 int4(as i32) -> dequant bf16, transposed [N][K].
    // Exact dequant: -8*s is an exponent shift (exact), fma rounds once.
    const int g = k0 / GROUP;
    const float4 sc = *reinterpret_cast<const float4*>(&S[(size_t)g * N_DIM + n0 + b_cq * 4]);
    const float bx = -8.f * sc.x, by = -8.f * sc.y, bz = -8.f * sc.z, bw = -8.f * sc.w;
    float fb[4][8];
#pragma unroll
    for (int k = 0; k < 8; ++k) {
      const int4 q = *reinterpret_cast<const int4*>(
          &Bq[(size_t)(k0 + b_ko + k) * N_DIM + n0 + b_cq * 4]);
      fb[0][k] = (float)q.x * sc.x + bx;   // v_cvt_f32_i32 + v_fma_f32
      fb[1][k] = (float)q.y * sc.y + by;
      fb[2][k] = (float)q.z * sc.z + bz;
      fb[3][k] = (float)q.w * sc.w + bw;
    }
#pragma unroll
    for (int c = 0; c < 4; ++c) {
      uint4 u;
      u.x = pack2_bf16(fb[c][0], fb[c][1]);
      u.y = pack2_bf16(fb[c][2], fb[c][3]);
      u.z = pack2_bf16(fb[c][4], fb[c][5]);
      u.w = pack2_bf16(fb[c][6], fb[c][7]);
      *reinterpret_cast<uint4*>(&sBt[buf][(b_cq * 4 + c) * LDB + b_ko]) = u;  // ds_store_b128
    }
  };

  stage(0, 0);

  for (int kt = 0; kt < KTILES; ++kt) {
    __syncthreads();  // tile kt visible; buf[(kt+1)&1] free for staging

    if (kt + 1 < KTILES) stage(kt + 1, (kt + 1) & 1);

    // ---------- compute: 2x8 WMMA tiles per wave from buf kt&1 ----------
    const int cur = kt & 1;
    const unsigned short* aBase = &sA[cur][(wm * 32 + frag_r) * LDA];
    const unsigned short* bBase = &sBt[cur][(wn * 128 + frag_r) * LDB];

    v16bf af[2];
#pragma unroll
    for (int mi = 0; mi < 2; ++mi)
      af[mi] = load_frag(aBase + (mi * 16) * LDA, frag_k0);

    // Register double-buffer the B fragment: load ni+1 while WMMAs for ni run,
    // so the pre-WMMA wait is dscnt<=2 instead of 0 (hides LDS latency).
    v16bf bcur = load_frag(bBase, frag_k0);
#pragma unroll
    for (int ni = 0; ni < 8; ++ni) {
      v16bf bnxt;
      if (ni < 7) bnxt = load_frag(bBase + ((ni + 1) * 16) * LDB, frag_k0);
      acc[0][ni] = __builtin_amdgcn_wmma_f32_16x16x32_bf16(
          false, af[0], false, bcur, (short)0, acc[0][ni], false, false);
      acc[1][ni] = __builtin_amdgcn_wmma_f32_16x16x32_bf16(
          false, af[1], false, bcur, (short)0, acc[1][ni], false, false);
      bcur = bnxt;
    }
  }

  // ---------- epilogue ----------
  // D layout: VGPR r of a 16x16 f32 tile: lanes 0-15 (M=r, N=lane), lanes 16-31 (M=8+r, N=lane-16).
  // Non-temporal stores: C is 192MB write-once; keep it from evicting Bq (192MB, reused by
  // all 32 M-blocks) out of the 192MB global L2.
  const int rb = m0 + wm * 32 + ((lane >> 4) << 3);
  const int cb = n0 + wn * 128 + (lane & 15);
#pragma unroll
  for (int mi = 0; mi < 2; ++mi) {
#pragma unroll
    for (int ni = 0; ni < 8; ++ni) {
#pragma unroll
      for (int r = 0; r < 8; ++r) {
        const int row = rb + mi * 16 + r;
        const int col = cb + ni * 16;
        __builtin_nontemporal_store(acc[mi][ni][r], &C[(size_t)row * N_DIM + col]);
      }
    }
  }
}

extern "C" void kernel_launch(void* const* d_in, const int* in_sizes, int n_in,
                              void* d_out, int out_size, void* d_ws, size_t ws_size,
                              hipStream_t stream) {
  (void)in_sizes; (void)n_in; (void)out_size; (void)d_ws; (void)ws_size;
  const float* a      = (const float*)d_in[0];
  const int*   b_q    = (const int*)d_in[1];
  const float* scales = (const float*)d_in[2];
  float*       out    = (float*)d_out;

  dim3 grid(M_DIM / BM, N_DIM / BN);  // 32 x 48
  dim3 block(256);                    // 8 wave32 waves
  machete_w4a32_bf16_wmma<<<grid, block, 0, stream>>>(a, b_q, scales, out);
}
